// LSTM_69758858822544
// MI455X (gfx1250) — compile-verified, matrix operand order checked by reading, not run on hardware
//
#include <hip/hip_runtime.h>
#include <stdint.h>

// ---------------------------------------------------------------------------
// LSTM  B=64, S=1024, F=512, H=512   (fused [x_t;h] @ [Wx;Wh], K = 1024)
// 32 persistent workgroups, each owns 16 h-columns (64 gate-columns).
// Weights: bf16 WMMA B-fragments resident in LDS (128 KB / WGP).
// A-matrix: bf16 fragments in global ws (x pre-converted; h double-buffered).
// ---------------------------------------------------------------------------

#define BATCH   64
#define SEQ     1024
#define FEAT    512
#define HID     512
#define NWG     32            // workgroups, each owns HID/NWG = 16 columns
#define NTHREADS 256          // 8 waves

typedef __attribute__((ext_vector_type(16))) __bf16 v16bf;
typedef __attribute__((ext_vector_type(8)))  float  v8f;

// round-to-nearest-even f32 -> bf16 (as raw u16)
__device__ __forceinline__ unsigned short f32_to_bf16(float f) {
  unsigned u = __float_as_uint(f);
  unsigned r = 0x7FFFu + ((u >> 16) & 1u);
  return (unsigned short)((u + r) >> 16);
}

// WMMA 16-bit A(16x32)/B(32x16) fragment layout (CDNA5 ISA 7.12.2):
// (lane, idx) -> (m_or_n, k) ; lanes 0-15 hold K {0..7,16..23}, 16-31 {8..15,24..31}
__device__ __forceinline__ void ab_layout(int lane, int idx, int& mn, int& kk) {
  mn = lane & 15;
  int hi = lane >> 4;
  int base = (idx < 8) ? idx : idx + 8;   // {0..7} or {16..23}
  kk = base + (hi << 3);
}

__device__ __forceinline__ float sigmoidf_(float x) {
  return 1.0f / (1.0f + __expf(-x));
}
__device__ __forceinline__ float tanhf_(float x) {
  return 1.0f - 2.0f / (__expf(2.0f * x) + 1.0f);
}

// ---------------------------------------------------------------------------
// Phase 0a: zero the grid-barrier counter and the h fragment buffers
// ---------------------------------------------------------------------------
__global__ void __launch_bounds__(256) lstm_init_ws(unsigned int* cnt,
                                                    unsigned short* hfrag) {
  int t = blockIdx.x * blockDim.x + threadIdx.x;
  if (t == 0) cnt[0] = 0u;
  for (int i = t; i < 2 * 64 * 512; i += gridDim.x * blockDim.x) hfrag[i] = 0;
}

// ---------------------------------------------------------------------------
// Phase 0b: convert x (B,S,F fp32) into bf16 A-fragment layout per step.
// Layout: xfrag[s][rowtile r(0..3)][ktile kt(0..15)][lane(0..31)][idx(0..15)]
// ---------------------------------------------------------------------------
__global__ void __launch_bounds__(256) lstm_xfrag(const float* __restrict__ x,
                                                  unsigned short* __restrict__ xfrag) {
  size_t e = ((size_t)blockIdx.x * 256 + threadIdx.x) * 4;
#pragma unroll
  for (int j = 0; j < 4; ++j, ++e) {
    int s    = (int)(e >> 15);          // 32768 elems per step
    int rem  = (int)(e & 32767);
    int frag = rem >> 9;
    int lane = (rem >> 4) & 31;
    int idx  = rem & 15;
    int r    = frag >> 4;               // row tile 0..3
    int kt   = frag & 15;               // k tile 0..15 (x half of K)
    int mm, kk; ab_layout(lane, idx, mm, kk);
    int b = r * 16 + mm;
    int f = kt * 32 + kk;
    float v = x[((size_t)b << 19) + ((size_t)s << 9) + (size_t)f]; // b*S*F + s*F + f
    xfrag[e] = f32_to_bf16(v);
  }
}

// ---------------------------------------------------------------------------
// Persistent recurrent kernel. 32 blocks x 256 threads.
// LDS: wfrag[4 gates][32 ktiles][512] bf16 (128KB) + preact[64][64] f32 (16KB)
//      + c_state[64][16] f32 (4KB)
// ---------------------------------------------------------------------------
__global__ void __launch_bounds__(NTHREADS, 1) lstm_persistent(
    const float* __restrict__ wii, const float* __restrict__ whi,
    const float* __restrict__ wif, const float* __restrict__ whf,
    const float* __restrict__ wig, const float* __restrict__ whg,
    const float* __restrict__ wio, const float* __restrict__ who,
    const float* __restrict__ bii, const float* __restrict__ bif,
    const float* __restrict__ big,
    const unsigned short* __restrict__ xfrag,
    unsigned short* hfrag, unsigned int* cnt, float* out) {
  extern __shared__ char smem_raw[];
  unsigned short* wfrag  = (unsigned short*)smem_raw;                   // 65536 u16
  float*          preact = (float*)(smem_raw + 131072);                 // 64*64 f32
  float*          cstate = (float*)(smem_raw + 131072 + 16384);         // 64*16 f32

  const int tid  = threadIdx.x;
  const int wg   = blockIdx.x;          // 0..31 -> columns [wg*16, wg*16+16)
  const int lane = tid & 31;
  const int wave = tid >> 5;
  const int r    = wave & 3;            // row tile of this wave
  const int ct0  = (wave >> 2) * 2;     // gates {ct0, ct0+1}

  const float* WX[4] = {wii, wif, wig, wio};
  const float* WH[4] = {whi, whf, whg, who};

  // ---- stage weights into LDS as pre-swizzled B fragments (once) ----
  for (int e = tid; e < 65536; e += NTHREADS) {
    int frag = e >> 9;
    int l    = (e >> 4) & 31;
    int idx  = e & 15;
    int ct   = frag >> 5;               // gate 0..3
    int kt   = frag & 31;               // k tile 0..31
    int n, kk; ab_layout(l, idx, n, kk);
    int k   = kt * 32 + kk;             // 0..1023
    int col = wg * 16 + n;              // 0..511
    float w = (k < FEAT) ? WX[ct][(size_t)k * HID + col]
                         : WH[ct][(size_t)(k - FEAT) * HID + col];
    wfrag[e] = f32_to_bf16(w);
  }
  for (int i = tid; i < 1024; i += NTHREADS) cstate[i] = 0.0f;
  __syncthreads();

  const unsigned int nwg = (unsigned int)gridDim.x;

  // loop-invariant base pointers (lane folded in; inner loops use imm offsets)
  const unsigned short* bp =
      wfrag + ((size_t)(ct0 * 32) << 9) + (size_t)lane * 16;          // LDS B frags
  const size_t lane16 = (size_t)lane * 16;
  const size_t rbase  = (size_t)r << 13;                              // r*16 frags *512

  for (int s = 0; s < SEQ; ++s) {
    const int cur = s & 1, nxt = cur ^ 1;
    v8f acc0 = {};
    v8f acc1 = {};

    const unsigned short* xp = xfrag + (((size_t)s) << 15) + rbase + lane16;
    const unsigned short* hp = hfrag + ((size_t)cur << 15) + rbase + lane16;

    // ---- x half of K (k tiles 0..15), all addressing via immediates ----
#pragma unroll
    for (int kt = 0; kt < 16; ++kt) {
      v16bf a  = *(const v16bf*)(xp + kt * 512);
      v16bf b0 = *(const v16bf*)(bp + kt * 512);
      v16bf b1 = *(const v16bf*)(bp + kt * 512 + 16384);
      acc0 = __builtin_amdgcn_wmma_f32_16x16x32_bf16(false, a, false, b0,
                                                     (short)0, acc0, false, false);
      acc1 = __builtin_amdgcn_wmma_f32_16x16x32_bf16(false, a, false, b1,
                                                     (short)0, acc1, false, false);
    }
    // ---- h half of K (k tiles 16..31) ----
#pragma unroll
    for (int kt = 0; kt < 16; ++kt) {
      v16bf a  = *(const v16bf*)(hp + kt * 512);
      v16bf b0 = *(const v16bf*)(bp + (16 + kt) * 512);
      v16bf b1 = *(const v16bf*)(bp + (16 + kt) * 512 + 16384);
      acc0 = __builtin_amdgcn_wmma_f32_16x16x32_bf16(false, a, false, b0,
                                                     (short)0, acc0, false, false);
      acc1 = __builtin_amdgcn_wmma_f32_16x16x32_bf16(false, a, false, b1,
                                                     (short)0, acc1, false, false);
    }

    // scatter C fragments (16x16 f32 layout) into preact LDS
    {
      int n  = lane & 15;
      int hi = lane >> 4;
#pragma unroll
      for (int reg = 0; reg < 8; ++reg) {
        int m = r * 16 + hi * 8 + reg;
        preact[m * 64 + ct0 * 16 + n]       = acc0[reg];
        preact[m * 64 + (ct0 + 1) * 16 + n] = acc1[reg];
      }
    }
    __syncthreads();

    // elementwise gate math + state update (faithful to reference quirks:
    // biases added twice; o-gate uses bif)
    for (int u = tid; u < 1024; u += NTHREADS) {
      int b    = u >> 4;
      int c16  = u & 15;
      int colg = wg * 16 + c16;
      float pi = preact[b * 64 + c16]      + 2.0f * bii[colg];
      float pf = preact[b * 64 + 16 + c16] + 2.0f * bif[colg];
      float pg = preact[b * 64 + 32 + c16] + 2.0f * big[colg];
      float po = preact[b * 64 + 48 + c16] + 2.0f * bif[colg];
      float it = sigmoidf_(pi);
      float ft = sigmoidf_(pf);
      float gt = tanhf_(pg);
      float ot = sigmoidf_(po);
      float cv = ft * cstate[u] + it * gt;
      cstate[u] = cv;
      float hv = ot * tanhf_(cv);

      // write h into fragment layout of hfrag[nxt]
      int kk  = colg & 31;
      int ktl = colg >> 5;                // 0..15
      int mm  = b & 15;
      int rr  = b >> 4;
      int hi2 = (kk >> 3) & 1;
      int l2  = mm + (hi2 << 4);
      int kk2 = kk - (hi2 << 3);
      int id2 = (kk2 < 8) ? kk2 : kk2 - 8;
      hfrag[((size_t)nxt << 15) + (size_t)((rr * 16 + ktl) << 9) + l2 * 16 + id2] =
          f32_to_bf16(hv);

      if (s == SEQ - 1) {
        out[b * HID + colg] = hv;                 // h_t
        out[BATCH * HID + b * HID + colg] = cv;   // c_t
      }
    }

    // ---- grid-wide barrier (monotonic counter) ----
    __threadfence();          // make this thread's h stores device-visible
    __syncthreads();          // whole WG flushed
    if (tid == 0) {
      atomicAdd(cnt, 1u);
      unsigned target = (unsigned)(s + 1) * nwg;
      while (atomicAdd(cnt, 0u) < target) { __builtin_amdgcn_s_sleep(2); }
    }
    __syncthreads();
    __threadfence();          // acquire: invalidate before reading next h buffer
  }
}

// ---------------------------------------------------------------------------
extern "C" void kernel_launch(void* const* d_in, const int* in_sizes, int n_in,
                              void* d_out, int out_size, void* d_ws, size_t ws_size,
                              hipStream_t stream) {
  const float* x   = (const float*)d_in[0];
  const float* wii = (const float*)d_in[1];
  const float* whi = (const float*)d_in[2];
  const float* wif = (const float*)d_in[3];
  const float* whf = (const float*)d_in[4];
  const float* wig = (const float*)d_in[5];
  const float* whg = (const float*)d_in[6];
  const float* wio = (const float*)d_in[7];
  const float* who = (const float*)d_in[8];
  const float* bii = (const float*)d_in[9];
  const float* bif = (const float*)d_in[10];
  const float* big = (const float*)d_in[11];
  float* out = (float*)d_out;

  unsigned char* ws = (unsigned char*)d_ws;
  unsigned int*   cnt   = (unsigned int*)ws;                               // 256 B
  unsigned short* xfrag = (unsigned short*)(ws + 256);                     // 64 MB
  unsigned short* hfrag = (unsigned short*)(ws + 256 + (size_t)67108864);  // 128 KB

  lstm_init_ws<<<32, 256, 0, stream>>>(cnt, hfrag);
  // 33,554,432 elements, 4 per thread
  lstm_xfrag<<<32768, 256, 0, stream>>>(x, xfrag);

  size_t smem = 131072 + 16384 + 4096;  // weights + preact + c state = 148 KB
  lstm_persistent<<<NWG, NTHREADS, smem, stream>>>(
      wii, whi, wif, whf, wig, whg, wio, who, bii, bif, big,
      xfrag, hfrag, cnt, out);
}
